// DeepseekV3DecoderLayer_24902220382976
// MI455X (gfx1250) — compile-verified, matrix operand order checked by reading, not run on hardware
//
#include <hip/hip_runtime.h>
#include <hip/hip_bf16.h>
#include <cstdint>

// ---------------- problem constants ----------------
#define Sdim 1024
#define Hdim 2048
#define QRr  1536
#define KVRr 512
#define RDd  64
#define NDd  128
#define VDd  128
#define NHh  16
#define QKDd 192
#define Ee   32
#define Ii   512
#define NSHh 2
#define NGg  8
#define KGg  2
#define TKk  8
#define GSs  4
#define Tt   1024

typedef __attribute__((ext_vector_type(16))) _Float16 v16h;
typedef __attribute__((ext_vector_type(8)))  float    v8f;

// ---------------- gfx1250 async global->LDS copy (ASYNCcnt path) ----------------
#if defined(__has_builtin)
# if __has_builtin(__builtin_amdgcn_global_load_async_to_lds_b128)
#  define HAVE_ASYNC_COPY 1
# endif
# if __has_builtin(__builtin_amdgcn_s_wait_asynccnt)
#  define HAVE_WAIT_ASYNC 1
# endif
#endif

#if defined(HAVE_ASYNC_COPY)
// Builtin signature (from clang diagnostic): param0 is
// '__attribute__((vector_size(16))) int __device__ *' -> AS(1) v4i*, LDS side AS(3).
typedef int v4i_t __attribute__((vector_size(16)));
typedef __attribute__((address_space(1))) v4i_t GlobalV4;
typedef __attribute__((address_space(3))) v4i_t LdsV4;
#endif

// Copy 16 bytes global -> LDS. Async (no VGPR round-trip) when available.
static __device__ __forceinline__ void copy16_g2l(const _Float16* gsrc, _Float16* ldst) {
#if defined(HAVE_ASYNC_COPY)
  __builtin_amdgcn_global_load_async_to_lds_b128((GlobalV4*)gsrc, (LdsV4*)ldst, 0, 0);
#else
  *(uint4*)ldst = *(const uint4*)gsrc;
#endif
}

static __device__ __forceinline__ void async_wait0() {
#if defined(HAVE_ASYNC_COPY)
# if defined(HAVE_WAIT_ASYNC)
  __builtin_amdgcn_s_wait_asynccnt(0);
# else
  asm volatile("s_wait_asynccnt 0" ::: "memory");
# endif
#endif
}

static __device__ __forceinline__ v8f wmma_f16(v16h a, v16h b, v8f c) {
  // D = A(16x32 f16) * B(32x16 f16) + C(16x16 f32)
  return __builtin_amdgcn_wmma_f32_16x16x32_f16(false, a, false, b, (short)0, c, false, false);
}

// A-layout fragment (16x32 tile, rows along ld, k contiguous).
// lane<16: row=lane, kb=0; lane>=16: kb=8; k in {kb..kb+7, kb+16..kb+23}.
static __device__ __forceinline__ v16h load_frag_rowk(const _Float16* p, int ld) {
  int lane = threadIdx.x & 31;
  const _Float16* q = p + (size_t)(lane & 15) * ld + ((lane >> 4) * 8);
  v16h f;
  *(uint4*)&f       = *(const uint4*)q;          // k = kb .. kb+7
  *((uint4*)&f + 1) = *(const uint4*)(q + 16);   // k = kb+16 .. kb+23
  return f;
}

// B fragment from [K][N]-major storage (32x16 tile): lane n fixed, k strided.
static __device__ __forceinline__ v16h load_frag_colk(const _Float16* p, int ld) {
  int lane = threadIdx.x & 31;
  int n = lane & 15, kb = (lane >> 4) * 8;
  v16h f;
#pragma unroll
  for (int i = 0; i < 4; ++i) {
    f[2*i]     = p[(size_t)(kb + 2*i)      * ld + n];
    f[2*i+1]   = p[(size_t)(kb + 2*i + 1)  * ld + n];
    f[8+2*i]   = p[(size_t)(kb + 16 + 2*i) * ld + n];
    f[8+2*i+1] = p[(size_t)(kb + 17 + 2*i) * ld + n];
  }
  return f;
}

// A fragment from an f32 row-major matrix (softmax probabilities in LDS).
static __device__ __forceinline__ v16h load_frag_rowk_f32(const float* p, int ld) {
  int lane = threadIdx.x & 31;
  const float* q = p + (size_t)(lane & 15) * ld + ((lane >> 4) * 8);
  v16h f;
#pragma unroll
  for (int i = 0; i < 8; ++i) { f[i] = (_Float16)q[i]; f[8+i] = (_Float16)q[16+i]; }
  return f;
}

// pack 4 f32 -> 4 f16 -> single 8B LDS store
static __device__ __forceinline__ void store4h(_Float16* d, float4 v) {
  union { _Float16 h[4]; unsigned long long u; } pk;
  pk.h[0] = (_Float16)v.x; pk.h[1] = (_Float16)v.y;
  pk.h[2] = (_Float16)v.z; pk.h[3] = (_Float16)v.w;
  *(unsigned long long*)d = pk.u;
}

// ---------------- generic GEMM: C[M,N] = A_f16[M,K] * W_f32[N,K]^T ----------------
// EPI: 0 = store, 1 = store + residual, 2 = atomicAdd
template<int EPI>
__global__ __launch_bounds__(256) void gemm_nk(
    const _Float16* __restrict__ A, const float* __restrict__ W,
    float* __restrict__ C, const float* __restrict__ Res,
    int M, int N, int K, int lda, int ldw, int ldc)
{
  __shared__ __align__(16) _Float16 As[128*72];
  __shared__ __align__(16) _Float16 Bs[128*72];
  const int LDT = 72;
  int m0 = blockIdx.y * 128, n0 = blockIdx.x * 128;
  int tid = threadIdx.x, lane = tid & 31, w = tid >> 5;
  int wm = (w & 1) * 64, wn = (w >> 1) * 32;
  v8f acc[4][2] = {};
  for (int k0 = 0; k0 < K; k0 += 64) {
#pragma unroll
    for (int it = 0; it < 4; ++it) {          // A tile: 128x64 halves, async DMA into LDS
      int idx = tid + it * 256;
      int r = idx >> 3, c8 = (idx & 7) * 8;
      int gm = m0 + r;
      if (gm < M) copy16_g2l(A + (size_t)gm * lda + k0 + c8, &As[r * LDT + c8]);
      else        *(uint4*)&As[r * LDT + c8] = make_uint4(0, 0, 0, 0);
    }
#pragma unroll
    for (int it = 0; it < 8; ++it) {          // W tile: 128 rows x 64 f32 -> f16 (VALU convert)
      int idx = tid + it * 256;
      int r = idx >> 4, c4 = (idx & 15) * 4;
      float4 v = make_float4(0.f, 0.f, 0.f, 0.f);
      int gn = n0 + r;
      if (gn < N) v = *(const float4*)(W + (size_t)gn * ldw + k0 + c4);
      store4h(&Bs[r * LDT + c4], v);
    }
    if (k0 + 64 < K) {
      __builtin_prefetch(A + (size_t)(m0 + (tid >> 1)) * lda + k0 + 64, 0, 1);
      __builtin_prefetch(W + (size_t)(n0 + (tid >> 1)) * ldw + k0 + 64, 0, 1);
    }
    async_wait0();
    __syncthreads();
#pragma unroll
    for (int kk = 0; kk < 64; kk += 32) {
      v16h b0 = load_frag_rowk(&Bs[(wn +  0) * LDT + kk], LDT);
      v16h b1 = load_frag_rowk(&Bs[(wn + 16) * LDT + kk], LDT);
#pragma unroll
      for (int i = 0; i < 4; ++i) {
        v16h a = load_frag_rowk(&As[(wm + i * 16) * LDT + kk], LDT);
        acc[i][0] = wmma_f16(a, b0, acc[i][0]);
        acc[i][1] = wmma_f16(a, b1, acc[i][1]);
      }
    }
    __syncthreads();
  }
  int cn = lane & 15, cm = (lane >> 4) * 8;
#pragma unroll
  for (int i = 0; i < 4; ++i)
#pragma unroll
    for (int j = 0; j < 2; ++j) {
      int gn = n0 + wn + j * 16 + cn;
      if (gn >= N) continue;
#pragma unroll
      for (int r = 0; r < 8; ++r) {
        int gm = m0 + wm + i * 16 + cm + r;
        if (gm >= M) continue;
        float v = acc[i][j][r];
        size_t o = (size_t)gm * ldc + gn;
        if (EPI == 1)      C[o] = v + Res[o];
        else if (EPI == 2) atomicAdd(&C[o], v);
        else               C[o] = v;
      }
    }
}

// ---------------- layernorm (f32 in -> f16 out) ----------------
__global__ __launch_bounds__(256) void layernorm_f16(
    const float* __restrict__ x, const float* __restrict__ w, const float* __restrict__ b,
    _Float16* __restrict__ out, int ncols, int ld_in, int ld_out)
{
  __shared__ float s1[256], s2[256];
  int row = blockIdx.x, tid = threadIdx.x;
  const float* xr = x + (size_t)row * ld_in;
  float a = 0.f, a2 = 0.f;
  for (int c = tid; c < ncols; c += 256) { float v = xr[c]; a += v; a2 += v * v; }
  s1[tid] = a; s2[tid] = a2; __syncthreads();
  for (int st = 128; st > 0; st >>= 1) {
    if (tid < st) { s1[tid] += s1[tid + st]; s2[tid] += s2[tid + st]; }
    __syncthreads();
  }
  float mean = s1[0] / ncols;
  float var  = s2[0] / ncols - mean * mean;
  float rstd = rsqrtf(var + 1e-5f);
  _Float16* orow = out + (size_t)row * ld_out;
  for (int c = tid; c < ncols; c += 256)
    orow[c] = (_Float16)((xr[c] - mean) * rstd * w[c] + b[c]);
}

// ---------------- rope + pack to head-major f16 (q pre-scaled by 1/sqrt(192)) ----------------
__global__ __launch_bounds__(256) void rope_pack(
    const float* __restrict__ qB, const float* __restrict__ kvB,
    const float* __restrict__ kvfull,
    _Float16* __restrict__ qf, _Float16* __restrict__ kf, _Float16* __restrict__ vf)
{
  int s = blockIdx.x, tid = threadIdx.x;
  __shared__ float cs[RDd], sn[RDd], kr[RDd];
  if (tid < RDd / 2) {
    float inv = powf(10000.f, -(float)(2 * tid) / (float)RDd);
    float fr = (float)s * inv;
    float c = cosf(fr), si = sinf(fr);
    cs[tid] = c; cs[tid + 32] = c; sn[tid] = si; sn[tid + 32] = si;
  }
  __syncthreads();
  if (tid < RDd) {
    const float* krl = kvfull + (size_t)s * (KVRr + RDd) + KVRr;
    float v = krl[tid];
    float rot = (tid < 32) ? -krl[tid + 32] : krl[tid - 32];
    kr[tid] = v * cs[tid] + rot * sn[tid];
  }
  __syncthreads();
  const float qscale = 0.07216878364870323f; // 1/sqrt(192)
  for (int idx = tid; idx < NHh * QKDd; idx += 256) {
    int h = idx / QKDd, d = idx % QKDd;
    const float* q = qB + (size_t)s * (NHh * QKDd) + (size_t)h * QKDd;
    float val;
    if (d < NDd) val = q[d];
    else {
      int r = d - NDd;
      float v = q[NDd + r];
      float rot = (r < 32) ? -q[NDd + r + 32] : q[NDd + r - 32];
      val = v * cs[r] + rot * sn[r];
    }
    qf[((size_t)h * Sdim + s) * QKDd + d] = (_Float16)(val * qscale);
    float kv;
    if (d < NDd) kv = kvB[(size_t)s * (NHh * (NDd + VDd)) + (size_t)h * (NDd + VDd) + d];
    else         kv = kr[d - NDd];
    kf[((size_t)h * Sdim + s) * QKDd + d] = (_Float16)kv;
  }
  for (int idx = tid; idx < NHh * VDd; idx += 256) {
    int h = idx / VDd, d = idx % VDd;
    vf[((size_t)h * Sdim + s) * VDd + d] =
      (_Float16)kvB[(size_t)s * (NHh * (NDd + VDd)) + (size_t)h * (NDd + VDd) + NDd + d];
  }
}

// ---------------- causal attention: 64 queries x full key strip in LDS ----------------
__global__ __launch_bounds__(256) void attn_kernel(
    const _Float16* __restrict__ qf, const _Float16* __restrict__ kf,
    const _Float16* __restrict__ vf, _Float16* __restrict__ ao)
{
  extern __shared__ float P[];          // [64][LDP]
  const int LDP = 1032;
  __shared__ float red[64][4];
  int h = blockIdx.y, q0 = blockIdx.x * 64;
  int tid = threadIdx.x, lane = tid & 31, w = tid >> 5;
  const _Float16* Q  = qf + (size_t)h * Sdim * QKDd;
  const _Float16* Km = kf + (size_t)h * Sdim * QKDd;
  const _Float16* V  = vf + (size_t)h * Sdim * VDd;
  int kend = q0 + 64;                   // exclusive causal bound
  int kb = w * 128;
  if (kb < kend) {                      // skip fully-masked key tiles
    int cn = lane & 15, cm = (lane >> 4) * 8;
    for (int qi = 0; qi < 4; ++qi) {
      const _Float16* Qp = Q + (size_t)(q0 + qi * 16) * QKDd;
      for (int j = 0; j < 8; ++j) {
        int kcol = kb + j * 16;
        if (kcol >= kend) break;
        v8f acc = {};
#pragma unroll
        for (int kk = 0; kk < QKDd; kk += 32) {
          v16h a = load_frag_rowk(Qp + kk, QKDd);
          v16h b = load_frag_rowk(Km + (size_t)kcol * QKDd + kk, QKDd);
          acc = wmma_f16(a, b, acc);
        }
#pragma unroll
        for (int r = 0; r < 8; ++r)
          P[(qi * 16 + cm + r) * LDP + kcol + cn] = acc[r];
      }
    }
  }
  __syncthreads();
  // softmax (4 threads per row)
  int row = tid >> 2, part = tid & 3;
  float* Pr = P + row * LDP;
  int kmax = q0 + row;                  // inclusive
  float mx = -1e30f;
  for (int k = part * 256; k < part * 256 + 256; ++k)
    if (k <= kmax) mx = fmaxf(mx, Pr[k]);
  red[row][part] = mx; __syncthreads();
  mx = fmaxf(fmaxf(red[row][0], red[row][1]), fmaxf(red[row][2], red[row][3]));
  float sum = 0.f;
  for (int k = part * 256; k < part * 256 + 256; ++k) {
    float e = (k <= kmax) ? __expf(Pr[k] - mx) : 0.f;
    Pr[k] = e; sum += e;
  }
  __syncthreads();
  red[row][part] = sum; __syncthreads();
  float tot = red[row][0] + red[row][1] + red[row][2] + red[row][3];
  float inv = 1.f / tot;
  for (int k = part * 256; k < part * 256 + 256; ++k) Pr[k] *= inv;
  __syncthreads();
  // PV: each wave -> one 16-query tile x 64 value cols
  int qi = w >> 1, nb = (w & 1) * 64;
  v8f acc[4] = {};
  for (int kk = 0; kk < kend; kk += 32) {
    v16h a = load_frag_rowk_f32(P + (size_t)(qi * 16) * LDP + kk, LDP);
#pragma unroll
    for (int j = 0; j < 4; ++j) {
      v16h b = load_frag_colk(V + (size_t)kk * VDd + nb + j * 16, VDd);
      acc[j] = wmma_f16(a, b, acc[j]);
    }
  }
  int cn = lane & 15, cm = (lane >> 4) * 8;
#pragma unroll
  for (int j = 0; j < 4; ++j)
#pragma unroll
    for (int r = 0; r < 8; ++r) {
      int s = q0 + qi * 16 + cm + r;
      ao[(size_t)s * (NHh * VDd) + h * VDd + nb + j * 16 + cn] = (_Float16)acc[j][r];
    }
}

// ---------------- gating ----------------
__global__ __launch_bounds__(256) void gate_logits(
    const _Float16* __restrict__ h2, const float* __restrict__ gw, float* __restrict__ logits)
{
  int t = blockIdx.x;
  int tid = threadIdx.x, lane = tid & 31, w = tid >> 5;
  int e = w * 4 + (lane >> 3), sub = lane & 7;
  const _Float16* hr = h2 + (size_t)t * Hdim;
  const float* g = gw + (size_t)e * Hdim;
  float s = 0.f;
  for (int c = sub; c < Hdim; c += 8) s += (float)hr[c] * g[c];
  s += __shfl_xor(s, 1, 8);
  s += __shfl_xor(s, 2, 8);
  s += __shfl_xor(s, 4, 8);
  if (sub == 0) logits[(size_t)t * Ee + e] = s;
}

__global__ void route_topk(const float* __restrict__ logits,
                           int* __restrict__ eidx, float* __restrict__ ew)
{
  int t = blockIdx.x * blockDim.x + threadIdx.x;
  if (t >= Tt) return;
  float lg[Ee];
  for (int i = 0; i < Ee; ++i) lg[i] = logits[(size_t)t * Ee + i];
  float gw[NGg * KGg]; int gi[NGg * KGg];
  for (int g = 0; g < NGg; ++g) {             // top-2 per group of 4
    int b1 = 0; float v1 = -1e30f;
    for (int j = 0; j < GSs; ++j) { float v = lg[g * GSs + j]; if (v > v1) { v1 = v; b1 = j; } }
    int b2 = 0; float v2 = -1e30f;
    for (int j = 0; j < GSs; ++j) { if (j == b1) continue; float v = lg[g * GSs + j]; if (v > v2) { v2 = v; b2 = j; } }
    gw[g * KGg] = v1; gi[g * KGg] = b1;
    gw[g * KGg + 1] = v2; gi[g * KGg + 1] = b2;
  }
  bool used[NGg * KGg] = {};
  float sum = 0.f;
  for (int s = 0; s < TKk; ++s) {             // top-8 of the 16 survivors
    int best = 0; float bv = -1e30f;
    for (int i = 0; i < NGg * KGg; ++i)
      if (!used[i] && gw[i] > bv) { bv = gw[i]; best = i; }
    used[best] = true;
    int grp = best / KGg;
    eidx[t * TKk + s] = grp * GSs + gi[best];
    ew[t * TKk + s] = bv; sum += bv;
  }
  float inv = 1.f / (sum + 1e-20f);
  for (int s = 0; s < TKk; ++s) ew[t * TKk + s] *= inv;
}

__global__ void assign_tokens(const int* __restrict__ eidx, const float* __restrict__ ew,
                              int* __restrict__ counts, int* __restrict__ lists,
                              float* __restrict__ lw)
{
  int t = blockIdx.x * blockDim.x + threadIdx.x;
  if (t >= Tt) return;
  for (int j = 0; j < TKk; ++j) {
    int e = eidx[t * TKk + j];
    int pos = atomicAdd(&counts[e], 1);
    lists[e * Tt + pos] = t;
    lw[e * Tt + pos] = ew[t * TKk + j];
  }
}

__global__ void init_ident(int* __restrict__ idlist, float* __restrict__ idw, int* __restrict__ shcnt)
{
  int t = blockIdx.x * blockDim.x + threadIdx.x;
  if (t < Tt) { idlist[t] = t; idw[t] = 1.f; }
  if (t == 0) shcnt[0] = Tt;
}

// ---------------- MoE: fused gate-up GEMM + SiLU*up (gathered tokens) ----------------
__global__ __launch_bounds__(256) void moe_gu_silu(
    const _Float16* __restrict__ h2, const float* __restrict__ gu_base, size_t gu_estride,
    const int* __restrict__ counts, const int* __restrict__ lists, int list_stride,
    _Float16* __restrict__ a_out, size_t a_estride)
{
  union SM { struct { _Float16 A[128 * 72]; _Float16 B[128 * 72]; } st; float c[128 * 132]; };
  __shared__ __align__(16) SM sm;
  const int LDT = 72;
  int e = blockIdx.z;
  int cnt = counts[e];
  int m0 = blockIdx.y * 128;
  if (m0 >= cnt) return;
  int p0 = blockIdx.x * 64;                                   // gate column base
  const float* Wg = gu_base + (size_t)e * gu_estride;         // [H][2I], ld = 2I
  const int* list = lists + (size_t)e * list_stride;
  int tid = threadIdx.x, lane = tid & 31, w = tid >> 5;
  int wm = (w & 1) * 64, wn = (w >> 1) * 32;
  v8f acc[4][2] = {};
  for (int k0 = 0; k0 < Hdim; k0 += 64) {
#pragma unroll
    for (int it = 0; it < 4; ++it) {                          // gather A rows (async DMA)
      int idx = tid + it * 256;
      int r = idx >> 3, c8 = (idx & 7) * 8;
      int rr = m0 + r;
      if (rr < cnt) {
        int tok = list[rr];
        copy16_g2l(h2 + (size_t)tok * Hdim + k0 + c8, &sm.st.A[r * LDT + c8]);
      } else {
        *(uint4*)&sm.st.A[r * LDT + c8] = make_uint4(0, 0, 0, 0);
      }
    }
#pragma unroll
    for (int it = 0; it < 8; ++it) {                          // W[K][N]: cols {p0..p0+63, p0+I..p0+I+63}
      int idx = tid + it * 256;
      int kr = idx >> 5, c4 = (idx & 31) * 4;
      int col = (c4 < 64) ? (p0 + c4) : (p0 + Ii + (c4 - 64));
      float4 v = *(const float4*)(Wg + (size_t)(k0 + kr) * (2 * Ii) + col);
      sm.st.B[(c4 + 0) * LDT + kr] = (_Float16)v.x;
      sm.st.B[(c4 + 1) * LDT + kr] = (_Float16)v.y;
      sm.st.B[(c4 + 2) * LDT + kr] = (_Float16)v.z;
      sm.st.B[(c4 + 3) * LDT + kr] = (_Float16)v.w;
    }
    async_wait0();
    __syncthreads();
#pragma unroll
    for (int kk = 0; kk < 64; kk += 32) {
      v16h b0 = load_frag_rowk(&sm.st.B[(wn + 0) * LDT + kk], LDT);
      v16h b1 = load_frag_rowk(&sm.st.B[(wn + 16) * LDT + kk], LDT);
#pragma unroll
      for (int i = 0; i < 4; ++i) {
        v16h a = load_frag_rowk(&sm.st.A[(wm + i * 16) * LDT + kk], LDT);
        acc[i][0] = wmma_f16(a, b0, acc[i][0]);
        acc[i][1] = wmma_f16(a, b1, acc[i][1]);
      }
    }
    __syncthreads();
  }
  int cn = lane & 15, cm = (lane >> 4) * 8;
#pragma unroll
  for (int i = 0; i < 4; ++i)
#pragma unroll
    for (int j = 0; j < 2; ++j)
#pragma unroll
      for (int r = 0; r < 8; ++r)
        sm.c[(wm + i * 16 + cm + r) * 132 + wn + j * 16 + cn] = acc[i][j][r];
  __syncthreads();
  for (int idx = tid; idx < 128 * 64; idx += 256) {
    int m = idx >> 6, j = idx & 63;
    if (m0 + m < cnt) {
      float g = sm.c[m * 132 + j];
      float u = sm.c[m * 132 + 64 + j];
      float av = (g / (1.f + __expf(-g))) * u;     // silu(g)*u
      a_out[(size_t)e * a_estride + (size_t)(m0 + m) * Ii + p0 + j] = (_Float16)av;
    }
  }
}

// ---------------- MoE: down projection with weighted scatter-add ----------------
__global__ __launch_bounds__(256) void moe_down_scatter(
    const _Float16* __restrict__ a_base, size_t a_estride,
    const float* __restrict__ down_base, size_t down_estride,
    const int* __restrict__ counts, const int* __restrict__ lists, int list_stride,
    const float* __restrict__ lw, int lw_stride,
    float* __restrict__ moeacc)
{
  __shared__ __align__(16) _Float16 As[128 * 72];
  __shared__ __align__(16) _Float16 Bs[128 * 72];
  const int LDT = 72;
  int e = blockIdx.z;
  int cnt = counts[e];
  int m0 = blockIdx.y * 128;
  if (m0 >= cnt) return;
  int n0 = blockIdx.x * 128;
  const _Float16* A = a_base + (size_t)e * a_estride;         // [cnt][I]
  const float* W = down_base + (size_t)e * down_estride;      // [I][H]
  const int* list = lists + (size_t)e * list_stride;
  const float* wrow = lw + (size_t)e * lw_stride;
  int tid = threadIdx.x, lane = tid & 31, w = tid >> 5;
  int wm = (w & 1) * 64, wn = (w >> 1) * 32;
  v8f acc[4][2] = {};
  for (int k0 = 0; k0 < Ii; k0 += 64) {
#pragma unroll
    for (int it = 0; it < 4; ++it) {                          // A tile (async DMA)
      int idx = tid + it * 256;
      int r = idx >> 3, c8 = (idx & 7) * 8;
      if (m0 + r < cnt) copy16_g2l(A + (size_t)(m0 + r) * Ii + k0 + c8, &As[r * LDT + c8]);
      else              *(uint4*)&As[r * LDT + c8] = make_uint4(0, 0, 0, 0);
    }
#pragma unroll
    for (int it = 0; it < 8; ++it) {                          // W[K][N], transpose into Bs[n][k]
      int idx = tid + it * 256;
      int kr = idx >> 5, c4 = (idx & 31) * 4;
      float4 v = *(const float4*)(W + (size_t)(k0 + kr) * Hdim + n0 + c4);
      Bs[(c4 + 0) * LDT + kr] = (_Float16)v.x;
      Bs[(c4 + 1) * LDT + kr] = (_Float16)v.y;
      Bs[(c4 + 2) * LDT + kr] = (_Float16)v.z;
      Bs[(c4 + 3) * LDT + kr] = (_Float16)v.w;
    }
    async_wait0();
    __syncthreads();
#pragma unroll
    for (int kk = 0; kk < 64; kk += 32) {
      v16h b0 = load_frag_rowk(&Bs[(wn + 0) * LDT + kk], LDT);
      v16h b1 = load_frag_rowk(&Bs[(wn + 16) * LDT + kk], LDT);
#pragma unroll
      for (int i = 0; i < 4; ++i) {
        v16h a = load_frag_rowk(&As[(wm + i * 16) * LDT + kk], LDT);
        acc[i][0] = wmma_f16(a, b0, acc[i][0]);
        acc[i][1] = wmma_f16(a, b1, acc[i][1]);
      }
    }
    __syncthreads();
  }
  int cn = lane & 15, cm = (lane >> 4) * 8;
#pragma unroll
  for (int i = 0; i < 4; ++i)
#pragma unroll
    for (int r = 0; r < 8; ++r) {
      int slot = m0 + wm + i * 16 + cm + r;
      if (slot >= cnt) continue;
      int tok = list[slot];
      float wt = wrow[slot];
#pragma unroll
      for (int j = 0; j < 2; ++j) {
        int gn = n0 + wn + j * 16 + cn;
        atomicAdd(&moeacc[(size_t)tok * Hdim + gn], wt * acc[i][j][r]);
      }
    }
}

__global__ void final_add(const float* __restrict__ xres, const float* __restrict__ moe,
                          float* __restrict__ out, int n)
{
  int i = blockIdx.x * blockDim.x + threadIdx.x;
  if (i < n) out[i] = xres[i] + moe[i];
}

// ---------------- host-side pipeline ----------------
extern "C" void kernel_launch(void* const* d_in, const int* in_sizes, int n_in,
                              void* d_out, int out_size, void* d_ws, size_t ws_size,
                              hipStream_t stream)
{
  (void)in_sizes; (void)n_in; (void)out_size; (void)ws_size;
  const float* x         = (const float*)d_in[0];
  const float* ln1_w     = (const float*)d_in[1];
  const float* ln1_b     = (const float*)d_in[2];
  const float* ln2_w     = (const float*)d_in[3];
  const float* ln2_b     = (const float*)d_in[4];
  const float* qkv_a_w   = (const float*)d_in[5];
  const float* q_a_ln_w  = (const float*)d_in[6];
  const float* q_a_ln_b  = (const float*)d_in[7];
  const float* q_b_w     = (const float*)d_in[8];
  const float* kv_a_w    = (const float*)d_in[9];
  const float* kv_a_ln_w = (const float*)d_in[10];
  const float* kv_a_ln_b = (const float*)d_in[11];
  const float* kv_b_w    = (const float*)d_in[12];
  const float* o_w       = (const float*)d_in[13];
  const float* gate_w    = (const float*)d_in[14];
  const float* expert_gu = (const float*)d_in[15];
  const float* expert_dn = (const float*)d_in[16];
  const float* shared_gu = (const float*)d_in[17];
  const float* shared_dn = (const float*)d_in[18];
  float* out = (float*)d_out;

  char* base = (char*)d_ws; size_t off = 0;
  auto alloc = [&](size_t bytes) -> void* {
    void* r = base + off;
    off += (bytes + 255) & ~(size_t)255;
    return r;
  };
  _Float16* h16    = (_Float16*)alloc((size_t)Sdim * Hdim * 2);
  float*    qkvA   = (float*)   alloc((size_t)Sdim * QRr * 4);
  float*    kvfull = (float*)   alloc((size_t)Sdim * (KVRr + RDd) * 4);
  _Float16* qlat16 = (_Float16*)alloc((size_t)Sdim * QRr * 2);
  _Float16* kvlat16= (_Float16*)alloc((size_t)Sdim * KVRr * 2);
  float*    qB     = (float*)   alloc((size_t)Sdim * NHh * QKDd * 4);
  float*    kvB    = (float*)   alloc((size_t)Sdim * NHh * (NDd + VDd) * 4);
  _Float16* qf16   = (_Float16*)alloc((size_t)NHh * Sdim * QKDd * 2);
  _Float16* kf16   = (_Float16*)alloc((size_t)NHh * Sdim * QKDd * 2);
  _Float16* vf16   = (_Float16*)alloc((size_t)NHh * Sdim * VDd * 2);
  _Float16* ao16   = (_Float16*)alloc((size_t)Sdim * NHh * VDd * 2);
  float*    xres   = (float*)   alloc((size_t)Sdim * Hdim * 4);
  _Float16* h216   = (_Float16*)alloc((size_t)Sdim * Hdim * 2);
  float*    logits = (float*)   alloc((size_t)Tt * Ee * 4);
  int*      eidx   = (int*)     alloc((size_t)Tt * TKk * 4);
  float*    ew     = (float*)   alloc((size_t)Tt * TKk * 4);
  int*      counts = (int*)     alloc((size_t)Ee * 4);
  int*      lists  = (int*)     alloc((size_t)Ee * Tt * 4);
  float*    lwts   = (float*)   alloc((size_t)Ee * Tt * 4);
  int*      idlist = (int*)     alloc((size_t)Tt * 4);
  float*    idw    = (float*)   alloc((size_t)Tt * 4);
  int*      shcnt  = (int*)     alloc(256);
  _Float16* aexp   = (_Float16*)alloc((size_t)Ee * Tt * Ii * 2);
  _Float16* ash    = (_Float16*)alloc((size_t)Tt * Ii * 2);
  float*    moeacc = (float*)   alloc((size_t)Sdim * Hdim * 4);

  // attention + projections
  layernorm_f16<<<Sdim, 256, 0, stream>>>(x, ln1_w, ln1_b, h16, Hdim, Hdim, Hdim);
  gemm_nk<0><<<dim3(QRr / 128, Sdim / 128), 256, 0, stream>>>(
      h16, qkv_a_w, qkvA, nullptr, Sdim, QRr, Hdim, Hdim, Hdim, QRr);
  gemm_nk<0><<<dim3((KVRr + RDd + 127) / 128, Sdim / 128), 256, 0, stream>>>(
      h16, kv_a_w, kvfull, nullptr, Sdim, KVRr + RDd, Hdim, Hdim, Hdim, KVRr + RDd);
  layernorm_f16<<<Sdim, 256, 0, stream>>>(qkvA, q_a_ln_w, q_a_ln_b, qlat16, QRr, QRr, QRr);
  layernorm_f16<<<Sdim, 256, 0, stream>>>(kvfull, kv_a_ln_w, kv_a_ln_b, kvlat16, KVRr, KVRr + RDd, KVRr);
  gemm_nk<0><<<dim3(NHh * QKDd / 128, Sdim / 128), 256, 0, stream>>>(
      qlat16, q_b_w, qB, nullptr, Sdim, NHh * QKDd, QRr, QRr, QRr, NHh * QKDd);
  gemm_nk<0><<<dim3(NHh * (NDd + VDd) / 128, Sdim / 128), 256, 0, stream>>>(
      kvlat16, kv_b_w, kvB, nullptr, Sdim, NHh * (NDd + VDd), KVRr, KVRr, KVRr, NHh * (NDd + VDd));
  rope_pack<<<Sdim, 256, 0, stream>>>(qB, kvB, kvfull, qf16, kf16, vf16);
  attn_kernel<<<dim3(Sdim / 64, NHh), 256, 64 * 1032 * sizeof(float), stream>>>(qf16, kf16, vf16, ao16);
  gemm_nk<1><<<dim3(Hdim / 128, Sdim / 128), 256, 0, stream>>>(
      ao16, o_w, xres, x, Sdim, Hdim, NHh * VDd, NHh * VDd, NHh * VDd, Hdim);

  // MoE
  layernorm_f16<<<Sdim, 256, 0, stream>>>(xres, ln2_w, ln2_b, h216, Hdim, Hdim, Hdim);
  gate_logits<<<Tt, 256, 0, stream>>>(h216, gate_w, logits);
  route_topk<<<(Tt + 255) / 256, 256, 0, stream>>>(logits, eidx, ew);
  (void)hipMemsetAsync(counts, 0, Ee * sizeof(int), stream);
  (void)hipMemsetAsync(moeacc, 0, (size_t)Sdim * Hdim * sizeof(float), stream);
  init_ident<<<(Tt + 255) / 256, 256, 0, stream>>>(idlist, idw, shcnt);
  assign_tokens<<<(Tt + 255) / 256, 256, 0, stream>>>(eidx, ew, counts, lists, lwts);

  moe_gu_silu<<<dim3(Ii / 64, Tt / 128, Ee), 256, 0, stream>>>(
      h216, expert_gu, (size_t)Hdim * 2 * Ii, counts, lists, Tt, aexp, (size_t)Tt * Ii);
  moe_down_scatter<<<dim3(Hdim / 128, Tt / 128, Ee), 256, 0, stream>>>(
      aexp, (size_t)Tt * Ii, expert_dn, (size_t)Ii * Hdim, counts, lists, Tt, lwts, Tt, moeacc);

  for (int sh = 0; sh < NSHh; ++sh) {
    moe_gu_silu<<<dim3(Ii / 64, Tt / 128, 1), 256, 0, stream>>>(
        h216, shared_gu + (size_t)sh * Hdim * 2 * Ii, 0, shcnt, idlist, 0, ash, 0);
    moe_down_scatter<<<dim3(Hdim / 128, Tt / 128, 1), 256, 0, stream>>>(
        ash, 0, shared_dn + (size_t)sh * Ii * Hdim, 0, shcnt, idlist, 0, idw, 0, moeacc);
  }

  final_add<<<(Sdim * Hdim + 255) / 256, 256, 0, stream>>>(xres, moeacc, out, Sdim * Hdim);
}
